// ModelParallelAMSoftmaxLoss_10746008174737
// MI455X (gfx1250) — matrix operand order repaired and checked by reading, not compiled, not run on hardware
//
#include <hip/hip_runtime.h>
#include <hip/hip_bf16.h>
#include <cstdint>

typedef __attribute__((ext_vector_type(16))) __bf16 v16bf;
typedef __attribute__((ext_vector_type(8)))  float  v8f;

#define AM_D      512
#define AM_SCALE  30.0f
#define AM_MARGIN 0.35f

// ---------- helpers ----------
static __device__ __forceinline__ unsigned int am_pack2_bf16(float a, float b) {
  unsigned int ua = __float_as_uint(a);
  unsigned int ub = __float_as_uint(b);
  ua = (ua + 0x7FFFu + ((ua >> 16) & 1u)) >> 16;   // RNE f32 -> bf16
  ub = (ub + 0x7FFFu + ((ub >> 16) & 1u)) >> 16;
  return ua | (ub << 16);
}

// ---------- kernel 1/2: row-normalize f32 -> bf16 (one wave per row of 512) ----------
__global__ void am_norm_bf16(const float* __restrict__ src,
                             unsigned short* __restrict__ dst, int rows) {
  int row  = blockIdx.x * 8 + (threadIdx.x >> 5);
  int lane = threadIdx.x & 31;
  if (row >= rows) return;
  const float4* p = (const float4*)(src + (size_t)row * AM_D);
  float4 v[4];
  float ss = 0.f;
#pragma unroll
  for (int i = 0; i < 4; ++i) {
    v[i] = p[lane + 32 * i];
    ss += v[i].x * v[i].x + v[i].y * v[i].y + v[i].z * v[i].z + v[i].w * v[i].w;
  }
#pragma unroll
  for (int m = 16; m >= 1; m >>= 1) ss += __shfl_xor(ss, m, 32);
  float inv = rsqrtf(ss + 1e-12f);
  uint2* q = (uint2*)(dst + (size_t)row * AM_D);
#pragma unroll
  for (int i = 0; i < 4; ++i) {
    uint2 o;
    o.x = am_pack2_bf16(v[i].x * inv, v[i].y * inv);
    o.y = am_pack2_bf16(v[i].z * inv, v[i].w * inv);
    q[lane + 32 * i] = o;
  }
}

// ---------- kernel 3: zero the per-row exp-sum accumulators ----------
__global__ void am_zero(float* __restrict__ S, int n) {
  int i = blockIdx.x * 256 + threadIdx.x;
  if (i < n) S[i] = 0.f;
}

// ---------- kernel 4: bf16 WMMA GEMM + streaming exp-sum ----------
// grid.x = class blocks of 512, grid.y = row blocks of 64; 256 threads (8 waves).
// Each wave: rows [n0, n0+64) as FOUR 16-row tiles, classes [c0, c0+64) as FOUR
// 16-col tiles -> 16 WMMAs per K-step against 8 fragment loads (32 FLOP/B).
__global__ void __launch_bounds__(256)
am_gemm_softmax(const unsigned short* __restrict__ xn,
                const unsigned short* __restrict__ wn,
                float* __restrict__ S, int C) {
  int lane = threadIdx.x & 31;
  int wave = threadIdx.x >> 5;
  int n0 = blockIdx.y * 64;
  int c0 = blockIdx.x * 512 + wave * 64;
  int l15 = lane & 15;
  int hiK = (lane & 16) ? 1 : 0;

  // A base (16-bit A 16x32 layout: lanes<16 hold K k0..k0+7 / k0+16..23).
  // Row tiles rt=0..3 sit at fixed strides of 16*AM_D elements (16 KB) -> immediates.
  const unsigned short* ap = xn + (size_t)(n0 + l15) * AM_D + (hiK ? 8 : 0);
  // B bases (16-bit B 32x16 layout: lane = column, lanes<16 hold K k0..k0+7 / k0+16..23)
  const unsigned short* bp[4];
#pragma unroll
  for (int f = 0; f < 4; ++f) {
    int cls = c0 + f * 16 + l15;
    if (cls > C - 1) cls = C - 1;        // clamp OOB loads; masked later
    bp[f] = wn + (size_t)cls * AM_D + (hiK ? 8 : 0);
  }

  v8f acc[4][4] = {};                    // [row tile][class tile], 128 VGPRs
  union AF { v16bf v; uint4 q[2]; };

  for (int k0 = 0; k0 < AM_D; k0 += 32) {
    AF a[4], b[4];
#pragma unroll
    for (int rt = 0; rt < 4; ++rt) {
      const unsigned short* arp = ap + (size_t)rt * 16 * AM_D;
      a[rt].q[0] = *(const uint4*)(arp + k0);
      a[rt].q[1] = *(const uint4*)(arp + k0 + 16);
    }
#pragma unroll
    for (int f = 0; f < 4; ++f) {
      b[f].q[0] = *(const uint4*)(bp[f] + k0);
      b[f].q[1] = *(const uint4*)(bp[f] + k0 + 16);
    }
#pragma unroll
    for (int f = 0; f < 4; ++f) {
#pragma unroll
      for (int rt = 0; rt < 4; ++rt) {
        acc[rt][f] = __builtin_amdgcn_wmma_f32_16x16x32_bf16(
            false, a[rt].v, false, b[f].v, (short)0, acc[rt][f], false, false);
      }
    }
  }

  // Streaming softmax partial: fixed shift of AM_SCALE (logits bounded: |30*cos|<=30).
  // C layout: VGPR r holds row M=r (lanes 0-15) / M=r+8 (lanes 16-31), col N = lane&15.
#pragma unroll
  for (int rt = 0; rt < 4; ++rt) {
#pragma unroll
    for (int r = 0; r < 8; ++r) {
      float sum = 0.f;
#pragma unroll
      for (int f = 0; f < 4; ++f) {
        int cls = c0 + f * 16 + l15;
        float e = __expf(AM_SCALE * acc[rt][f][r] - AM_SCALE);
        sum += (cls < C) ? e : 0.f;
      }
      sum += __shfl_xor(sum, 1, 32);   // reduce across the 16-lane column group
      sum += __shfl_xor(sum, 2, 32);
      sum += __shfl_xor(sum, 4, 32);
      sum += __shfl_xor(sum, 8, 32);
      if (l15 == r) {
        int row = n0 + rt * 16 + r + ((lane >> 4) << 3);
        atomicAdd(&S[row], sum);
      }
    }
  }
}

// ---------- kernel 5: target cosine per sample (bf16 dot, consistent with GEMM) ----------
__global__ void am_target(const unsigned short* __restrict__ xn,
                          const unsigned short* __restrict__ wn,
                          const int* __restrict__ lb,
                          float* __restrict__ cosT, int N, int C) {
  int row  = blockIdx.x * 8 + (threadIdx.x >> 5);
  int lane = threadIdx.x & 31;
  if (row >= N) return;
  int t = lb[row];
  if (t < 0) t = 0;
  if (t >= C) t = C - 1;
  const uint4* xp = (const uint4*)(xn + (size_t)row * AM_D);
  const uint4* wp = (const uint4*)(wn + (size_t)t * AM_D);
  float acc = 0.f;
#pragma unroll
  for (int i = 0; i < 2; ++i) {
    uint4 a = xp[lane + 32 * i];
    uint4 b = wp[lane + 32 * i];
    const unsigned int* aq = (const unsigned int*)&a;
    const unsigned int* bq = (const unsigned int*)&b;
#pragma unroll
    for (int j = 0; j < 4; ++j) {
      float ax = __uint_as_float(aq[j] << 16);
      float ay = __uint_as_float(aq[j] & 0xFFFF0000u);
      float bx = __uint_as_float(bq[j] << 16);
      float by = __uint_as_float(bq[j] & 0xFFFF0000u);
      acc = fmaf(ax, bx, acc);
      acc = fmaf(ay, by, acc);
    }
  }
#pragma unroll
  for (int m = 16; m >= 1; m >>= 1) acc += __shfl_xor(acc, m, 32);
  if (lane == 0) cosT[row] = acc;
}

// ---------- kernel 6: margin fixup + log + mean ----------
__global__ void am_loss(const float* __restrict__ S, const float* __restrict__ cosT,
                        float* __restrict__ out, int N) {
  __shared__ float red[256];
  int tid = threadIdx.x;
  float local = 0.f;
  for (int n = tid; n < N; n += 256) {
    float c  = cosT[n];
    float tl = AM_SCALE * (c - AM_MARGIN);
    float Sfix = S[n] - __expf(AM_SCALE * c - AM_SCALE) + __expf(tl - AM_SCALE);
    // -logp = log(Sfix) + shift - target_logit   (shift = AM_SCALE)
    local += logf(Sfix) + AM_SCALE - tl;
  }
  red[tid] = local;
  __syncthreads();
  for (int s = 128; s > 0; s >>= 1) {
    if (tid < s) red[tid] += red[tid + s];
    __syncthreads();
  }
  if (tid == 0) out[0] = red[0] / (float)N;
}

// ---------- launch ----------
extern "C" void kernel_launch(void* const* d_in, const int* in_sizes, int n_in,
                              void* d_out, int out_size, void* d_ws, size_t ws_size,
                              hipStream_t stream) {
  const float* x  = (const float*)d_in[0];
  const float* W  = (const float*)d_in[1];
  const int*   lb = (const int*)d_in[2];
  float* out = (float*)d_out;

  int N = in_sizes[0] / AM_D;   // 2048
  int C = in_sizes[1] / AM_D;   // 100000

  char* ws = (char*)d_ws;
  size_t off = 0;
  unsigned short* xn = (unsigned short*)(ws + off);
  off += (size_t)N * AM_D * sizeof(unsigned short);
  off = (off + 255) & ~(size_t)255;
  unsigned short* wn = (unsigned short*)(ws + off);
  off += (size_t)C * AM_D * sizeof(unsigned short);
  off = (off + 255) & ~(size_t)255;
  float* S = (float*)(ws + off);
  off += (size_t)N * sizeof(float);
  off = (off + 255) & ~(size_t)255;
  float* cosT = (float*)(ws + off);

  am_norm_bf16<<<dim3((N + 7) / 8), 256, 0, stream>>>(x, xn, N);
  am_norm_bf16<<<dim3((C + 7) / 8), 256, 0, stream>>>(W, wn, C);
  am_zero<<<dim3((N + 255) / 256), 256, 0, stream>>>(S, N);

  int cblocks = (C + 511) / 512;       // 196
  int rblocks = N / 64;                // 32
  am_gemm_softmax<<<dim3(cblocks, rblocks), 256, 0, stream>>>(xn, wn, S, C);

  am_target<<<dim3((N + 7) / 8), 256, 0, stream>>>(xn, wn, lb, cosT, N, C);
  am_loss<<<1, 256, 0, stream>>>(S, cosT, out, N);
}
